// QuantizedLinear_34832184770789
// MI455X (gfx1250) — matrix operand order repaired and checked by reading, not compiled, not run on hardware
//
#include <hip/hip_runtime.h>

typedef __attribute__((ext_vector_type(16))) _Float16 v16h;
typedef __attribute__((ext_vector_type(8)))  _Float16 v8h;
typedef __attribute__((ext_vector_type(4)))  _Float16 v4h;
typedef __attribute__((ext_vector_type(8)))  float    v8f;

constexpr int Bc = 4, Sc = 2048, Kc = 4096, Nc = 11008;
constexpr int Mc = Bc * Sc;                 // 8192 rows of x

#define TILE_M 128
#define TILE_N 128
#define KSTEP  32
#define LDSPAD 8
#define LDS_STRIDE (KSTEP + LDSPAD)         // 40 halves = 80 bytes/row

union Frag16 { v16h v; v8h h[2]; };

__global__ __launch_bounds__(256)
void qlinear_wmma_f16(const float* __restrict__ x,
                      const int*   __restrict__ qw,
                      const float* __restrict__ scale,
                      const float* __restrict__ zp,
                      const float* __restrict__ bias,
                      float*       __restrict__ out)
{
    __shared__ _Float16 lx[2][TILE_M][LDS_STRIDE];   // x tile, f16
    __shared__ _Float16 lw[2][TILE_N][LDS_STRIDE];   // dequantized weight tile, f16

    const int tid   = threadIdx.x;
    const int mBase = blockIdx.y * TILE_M;
    const int nBase = blockIdx.x * TILE_N;

    // ---- staging assignment: each thread owns 16 contiguous K-elems of one row ----
    const int srow = tid >> 1;              // 0..127
    const int scol = (tid & 1) * 16;        // 0 or 16
    const float* xrow = x  + (size_t)(mBase + srow) * Kc + scol;
    const int*   qrow = qw + (size_t)(nBase + srow) * Kc + scol;
    const float  s = scale[nBase + srow];   // per-output-channel, loop-invariant
    const float  z = zp[nBase + srow];

    // ---- wave / fragment geometry (wave32) ----
    const int lane = tid & 31;
    const int r    = lane & 15;
    const int hs   = lane >> 4;             // half-select: lanes 0-15 vs 16-31
    const int wave = tid >> 5;              // 0..7
    const int wm   = wave & 3;              // 4 slabs of 32 rows
    const int wn   = wave >> 2;             // 2 slabs of 64 cols

    v8f acc[2][4] = {};

    auto stage = [&](int buf, int k0) {
        const float4* xf = (const float4*)(xrow + k0);
        const int4*   qf = (const int4*)(qrow + k0);
        #pragma unroll
        for (int c = 0; c < 4; ++c) {
            float4 fx = xf[c];
            int4   qi = qf[c];
            v4h hx = { (_Float16)fx.x, (_Float16)fx.y,
                       (_Float16)fx.z, (_Float16)fx.w };
            v4h hw = { (_Float16)(((float)qi.x - z) * s),
                       (_Float16)(((float)qi.y - z) * s),
                       (_Float16)(((float)qi.z - z) * s),
                       (_Float16)(((float)qi.w - z) * s) };
            *(v4h*)&lx[buf][srow][scol + c * 4] = hx;
            *(v4h*)&lw[buf][srow][scol + c * 4] = hw;
        }
    };

    stage(0, 0);
    __syncthreads();

    constexpr int NKT = Kc / KSTEP;         // 128 K-steps
    for (int kt = 0; kt < NKT; ++kt) {
        const int cur = kt & 1;
        if (kt + 1 < NKT) stage(cur ^ 1, (kt + 1) * KSTEP);

        // A fragments: 16x32 f16, lane layout per ISA 7.12.2
        Frag16 a[2], b[4];
        #pragma unroll
        for (int i = 0; i < 2; ++i) {
            const _Float16* ap = &lx[cur][wm * 32 + i * 16 + r][0];
            a[i].h[0] = *(const v8h*)(ap + hs * 8);        // K 0..7 / 8..15
            a[i].h[1] = *(const v8h*)(ap + 16 + hs * 8);   // K 16..23 / 24..31
        }
        // B fragments: 32x16 f16, lane holds one weight row, K 0..7/8..15 then 16..23/24..31
        #pragma unroll
        for (int j = 0; j < 4; ++j) {
            const _Float16* bp = &lw[cur][wn * 64 + j * 16 + r][hs * 8];
            b[j].h[0] = *(const v8h*)(bp);
            b[j].h[1] = *(const v8h*)(bp + 16);
        }

        #pragma unroll
        for (int i = 0; i < 2; ++i)
            #pragma unroll
            for (int j = 0; j < 4; ++j)
                acc[i][j] = __builtin_amdgcn_wmma_f32_16x16x32_f16(
                    false, a[i].v, false, b[j].v,
                    (short)0, acc[i][j], false, false);

        __syncthreads();
    }

    // ---- epilogue: bias add in f32, store per ISA C/D layout ----
    #pragma unroll
    for (int j = 0; j < 4; ++j) {
        const int n  = nBase + wn * 64 + j * 16 + r;
        const float bb = bias[n];
        #pragma unroll
        for (int i = 0; i < 2; ++i) {
            const int mrow = mBase + wm * 32 + i * 16 + hs * 8;
            #pragma unroll
            for (int v = 0; v < 8; ++v)
                out[(size_t)(mrow + v) * Nc + n] = acc[i][j][v] + bb;
        }
    }
}

extern "C" void kernel_launch(void* const* d_in, const int* in_sizes, int n_in,
                              void* d_out, int out_size, void* d_ws, size_t ws_size,
                              hipStream_t stream) {
    const float* x     = (const float*)d_in[0];
    const int*   qw    = (const int*)  d_in[1];
    const float* scale = (const float*)d_in[2];
    const float* zp    = (const float*)d_in[3];
    const float* bias  = (const float*)d_in[4];
    float*       out   = (float*)d_out;

    dim3 grid(Nc / TILE_N, Mc / TILE_M);   // (86, 64)
    dim3 block(256);                       // 8 wave32 waves per WGP workgroup
    hipLaunchKernelGGL(qlinear_wmma_f16, grid, block, 0, stream,
                       x, qw, scale, zp, bias, out);
}
